// GAT_20461224198764
// MI455X (gfx1250) — matrix operand order repaired and checked by reading, not compiled, not run on hardware
//
#include <hip/hip_runtime.h>
#include <cstdint>
#include <cstddef>

typedef __attribute__((ext_vector_type(16))) __bf16 v16bf;
typedef __attribute__((ext_vector_type(8)))  float  v8f;

#define NEG_SLOPE 0.2f

// ---------- ordered-float <-> uint key (for atomicMax on floats incl. negatives) ----------
__device__ __forceinline__ unsigned f32key(float f) {
  unsigned u = __float_as_uint(f);
  return (u & 0x80000000u) ? ~u : (u | 0x80000000u);
}
__device__ __forceinline__ float key2f32(unsigned k) {
  unsigned u = (k & 0x80000000u) ? (k ^ 0x80000000u) : ~k;
  return __uint_as_float(u);
}

// ---------- fill kernels (workspace init each call; ws is poisoned once) ----------
__global__ void k_fill_f32(float* __restrict__ p, float v, int n) {
  int i = blockIdx.x * blockDim.x + threadIdx.x;
  if (i < n) p[i] = v;
}
__global__ void k_fill_u32(unsigned* __restrict__ p, unsigned v, int n) {
  int i = blockIdx.x * blockDim.x + threadIdx.x;
  if (i < n) p[i] = v;
}

// ---------- pack W[K,NC] (f32, row-major) into bf16 WMMA B-fragment order ----------
// P[((kb*2 + hi)*NC + col)*16 + e] = bf16(W[k,col]),  k = kb*32 + hi*16 + e
// so each lane's B fragment is one contiguous 32B block -> 2x global_load_b128, no cvt.
__global__ void k_pack_w(const float* __restrict__ W, __bf16* __restrict__ P, int K, int NC) {
  int idx = blockIdx.x * blockDim.x + threadIdx.x;
  if (idx >= K * NC) return;
  int k = idx / NC, col = idx - k * NC;
  int kb = k >> 5, kk = k & 31, hi = kk >> 4, e = kk & 15;
  P[(((size_t)(kb * 2 + hi) * NC + col) << 4) + e] = (__bf16)W[idx];
}

// ---------- bf16 WMMA GEMM: Out[M,NC] = A[M,K] @ W[K,NC], fp32 acc ----------
// WAVES waves per block, each wave owns CPW 16x16 output tiles sharing one A fragment.
// A tile (16 x K, f32) is staged into LDS with GLOBAL_LOAD_ASYNC_TO_LDS_B128 (ASYNCcnt),
// converted to bf16 at fragment-build time. B comes pre-packed in fragment order.
template<int K, int WAVES, int CPW>
__global__ __launch_bounds__(WAVES * 32)
void k_gemm_bf16(const float* __restrict__ A, const __bf16* __restrict__ Wp,
                 float* __restrict__ Out, int M) {
  constexpr int NC = WAVES * CPW * 16;
  constexpr int KP = K + 4;                 // +16B pad: keeps rows 16B-aligned, banks spread
  __shared__ float Asf[16 * KP];

  const int tid  = threadIdx.x;
  const int nthr = WAVES * 32;
  const int row0 = blockIdx.x * 16;

  // async DMA the 16xK f32 A tile into LDS (no VGPR round trip).
  // nf4 is an exact multiple of nthr for our instantiations -> EXEC full on every issue.
  const int nf4 = 16 * (K / 4);
  for (int i = tid; i < nf4; i += nthr) {
    int r  = i / (K / 4);
    int c4 = i - r * (K / 4);
    int row = row0 + r; if (row >= M) row = M - 1;   // clamp (safe re-read)
    const float* gp = A + (size_t)row * K + (size_t)c4 * 4;
    unsigned lp = (unsigned)(uintptr_t)(&Asf[r * KP + c4 * 4]);
    asm volatile("global_load_async_to_lds_b128 %0, %1, off"
                 :: "v"(lp), "v"(gp) : "memory");
  }
  asm volatile("s_wait_asynccnt 0x0" ::: "memory");
  __syncthreads();

  const int wave = tid >> 5;
  const int lane = tid & 31;
  const int m    = lane & 15;
  const int hi   = lane >> 4;
  const int col0 = wave * (CPW * 16) + m;   // first B/D column for this lane

  v8f acc[CPW] = {};
  for (int k0 = 0; k0 < K; k0 += 32) {
    const int kb2 = (k0 >> 5) * 2 + hi;
    // hint next packed-B block toward the caches (emits global_prefetch_b8)
    const int kb2n = (((k0 + 32 < K) ? (k0 + 32) : k0) >> 5) * 2 + hi;
    __builtin_prefetch(Wp + (((size_t)kb2n * NC + col0) << 4), 0, 3);

    v16bf a;
#pragma unroll
    for (int e = 0; e < 16; ++e) {
      const int ak = hi * 16 + e;  // A k = 16*hi + e (matches packed B)
      a[e] = (__bf16)Asf[m * KP + k0 + ak];
    }
#pragma unroll
    for (int t = 0; t < CPW; ++t) {
      const int col = col0 + t * 16;
      v16bf b = *reinterpret_cast<const v16bf*>(Wp + (((size_t)kb2 * NC + col) << 4));
      acc[t] = __builtin_amdgcn_wmma_f32_16x16x32_bf16(false, a, false, b,
                                                       (short)0, acc[t], false, false);
    }
  }

  // D: VGPR r -> row = r + 8*hi, col = col0 + t*16
  if (row0 + 16 <= M) {                      // whole-tile fast path: straight-line stores
#pragma unroll
    for (int t = 0; t < CPW; ++t) {
#pragma unroll
      for (int r = 0; r < 8; ++r)
        Out[(size_t)(row0 + r + 8 * hi) * NC + col0 + t * 16] = acc[t][r];
    }
  } else {
#pragma unroll
    for (int t = 0; t < CPW; ++t) {
#pragma unroll
      for (int r = 0; r < 8; ++r) {
        int row = row0 + r + 8 * hi;
        if (row < M) Out[(size_t)row * NC + col0 + t * 16] = acc[t][r];
      }
    }
  }
}

// ---------- per-node attention logits: as[n,h] = <h[n,h,:], att_s[h,:]> ----------
__global__ void k_alpha(const float* __restrict__ hfeat, const float* __restrict__ att_s,
                        const float* __restrict__ att_d, float* __restrict__ as_,
                        float* __restrict__ ad_, int N, int H, int C) {
  int idx = blockIdx.x * blockDim.x + threadIdx.x;
  if (idx >= N * H) return;
  int n = idx / H, hh = idx - n * H;
  const float* hp = hfeat + (size_t)n * H * C + (size_t)hh * C;
  const float* sp = att_s + hh * C;
  const float* dp = att_d + hh * C;
  float ds = 0.f, dd = 0.f;
  for (int c = 0; c < C; ++c) { float v = hp[c]; ds += v * sp[c]; dd += v * dp[c]; }
  as_[idx] = ds; ad_[idx] = dd;
}

// ---------- edge pass 1: segment max over dst (ordered-key atomicMax) ----------
__global__ void k_edge_max(const int* __restrict__ src, const int* __restrict__ dst,
                           int E, int N, int H,
                           const float* __restrict__ as_, const float* __restrict__ ad_,
                           unsigned* __restrict__ mkey) {
  int total = E + N;
  for (int e = blockIdx.x * blockDim.x + threadIdx.x; e < total;
       e += gridDim.x * blockDim.x) {
    int s, d;
    if (e < E) { s = src[e]; d = dst[e]; } else { s = d = e - E; }  // self-loops
    for (int h = 0; h < H; ++h) {
      float v = as_[(size_t)s * H + h] + ad_[(size_t)d * H + h];
      v = (v > 0.f) ? v : NEG_SLOPE * v;
      atomicMax(&mkey[(size_t)d * H + h], f32key(v));
    }
  }
}

// ---------- edge pass 2: segment sum of exp(e - m[dst]) ----------
__global__ void k_edge_sum(const int* __restrict__ src, const int* __restrict__ dst,
                           int E, int N, int H,
                           const float* __restrict__ as_, const float* __restrict__ ad_,
                           const unsigned* __restrict__ mkey, float* __restrict__ ssum) {
  int total = E + N;
  for (int e = blockIdx.x * blockDim.x + threadIdx.x; e < total;
       e += gridDim.x * blockDim.x) {
    int s, d;
    if (e < E) { s = src[e]; d = dst[e]; } else { s = d = e - E; }
    for (int h = 0; h < H; ++h) {
      float v = as_[(size_t)s * H + h] + ad_[(size_t)d * H + h];
      v = (v > 0.f) ? v : NEG_SLOPE * v;
      float m = key2f32(mkey[(size_t)d * H + h]);
      atomicAdd(&ssum[(size_t)d * H + h], __expf(v - m));
    }
  }
}

// ---------- edge pass 3: weighted scatter-add of messages, one thread per (edge, head) ----------
__global__ void k_edge_agg(const int* __restrict__ src, const int* __restrict__ dst,
                           int E, int N, int H, int C,
                           const float* __restrict__ as_, const float* __restrict__ ad_,
                           const unsigned* __restrict__ mkey, const float* __restrict__ ssum,
                           const float* __restrict__ hfeat, float* __restrict__ agg) {
  int total = (E + N) * H;
  for (int idx = blockIdx.x * blockDim.x + threadIdx.x; idx < total;
       idx += gridDim.x * blockDim.x) {
    int e = idx / H, h = idx - e * H;
    int s, d;
    if (e < E) { s = src[e]; d = dst[e]; } else { s = d = e - E; }
    float v = as_[(size_t)s * H + h] + ad_[(size_t)d * H + h];
    v = (v > 0.f) ? v : NEG_SLOPE * v;
    float m = key2f32(mkey[(size_t)d * H + h]);
    float alpha = __expf(v - m) / (ssum[(size_t)d * H + h] + 1e-16f);
    const float* hp = hfeat + ((size_t)s * H + h) * C;
    float* op = agg + ((size_t)d * H + h) * C;
    for (int c = 0; c < C; ++c) atomicAdd(&op[c], hp[c] * alpha);
  }
}

// ---------- bias + relu ----------
__global__ void k_bias_relu(const float* __restrict__ agg, const float* __restrict__ bias,
                            float* __restrict__ out, int N, int F) {
  int idx = blockIdx.x * blockDim.x + threadIdx.x;
  if (idx >= N * F) return;
  float v = agg[idx] + bias[idx % F];
  out[idx] = fmaxf(v, 0.f);
}

// ---------- final: bias + log_softmax over 64 channels; one wave32 per node ----------
__global__ __launch_bounds__(256)
void k_logsoftmax64(const float* __restrict__ agg, const float* __restrict__ bias,
                    float* __restrict__ out, int N) {
  int wave = threadIdx.x >> 5, lane = threadIdx.x & 31;
  int n = blockIdx.x * 8 + wave;
  if (n >= N) return;
  float v0 = agg[(size_t)n * 64 + lane]      + bias[lane];
  float v1 = agg[(size_t)n * 64 + 32 + lane] + bias[32 + lane];
  float mx = fmaxf(v0, v1);
  for (int off = 16; off; off >>= 1) mx = fmaxf(mx, __shfl_xor(mx, off, 32));
  float se = __expf(v0 - mx) + __expf(v1 - mx);
  for (int off = 16; off; off >>= 1) se += __shfl_xor(se, off, 32);
  float l = mx + __logf(se);
  out[(size_t)n * 64 + lane]      = v0 - l;
  out[(size_t)n * 64 + 32 + lane] = v1 - l;
}

extern "C" void kernel_launch(void* const* d_in, const int* in_sizes, int n_in,
                              void* d_out, int out_size, void* d_ws, size_t ws_size,
                              hipStream_t stream) {
  const float* x    = (const float*)d_in[0];
  const int*   ei   = (const int*)d_in[1];
  const float* W1   = (const float*)d_in[2];
  const float* aS1  = (const float*)d_in[3];
  const float* aD1  = (const float*)d_in[4];
  const float* b1   = (const float*)d_in[5];
  const float* W2   = (const float*)d_in[6];
  const float* aS2  = (const float*)d_in[7];
  const float* aD2  = (const float*)d_in[8];
  const float* b2   = (const float*)d_in[9];
  float* out = (float*)d_out;

  const int Fin = 512, H1 = 8, C1 = 16, F1 = 128, F2 = 64;
  const int N = in_sizes[0] / Fin;
  const int E = in_sizes[1] / 2;
  const int* src = ei;
  const int* dst = ei + E;

  // workspace layout (with aliasing: h1 region -> act1; agg1 region -> h2)
  float* w = (float*)d_ws;
  float* h1   = w;              w += (size_t)N * F1;   // layer-1 features, later act1
  float* agg1 = w;              w += (size_t)N * F1;   // layer-1 aggregate, later h2
  float* as1  = w;              w += (size_t)N * H1;
  float* ad1  = w;              w += (size_t)N * H1;
  unsigned* m1 = (unsigned*)w;  w += (size_t)N * H1;
  float* s1   = w;              w += (size_t)N * H1;
  float* agg2 = w;              w += (size_t)N * F2;
  float* as2  = w;              w += N;
  float* ad2  = w;              w += N;
  unsigned* m2 = (unsigned*)w;  w += N;
  float* s2   = w;              w += N;
  // packed bf16 weight tables (32B aligned)
  uintptr_t wp = (((uintptr_t)w) + 31u) & ~(uintptr_t)31u;
  __bf16* wpk1 = (__bf16*)wp;                          // 512*128 bf16
  __bf16* wpk2 = (__bf16*)(wp + (size_t)Fin * F1 * 2); // 128*64 bf16

  const unsigned NEGINF_KEY = 0x007FFFFFu;  // f32key(-inf)
  const int BLK = 256;
  const int totE = E + N;
  const int egrid = (totE + BLK - 1) / BLK;

  // pack both weight matrices into WMMA B-fragment order (tiny, once per call)
  k_pack_w<<<(Fin * F1 + BLK - 1) / BLK, BLK, 0, stream>>>(W1, wpk1, Fin, F1);
  k_pack_w<<<(F1 * F2 + BLK - 1) / BLK, BLK, 0, stream>>>(W2, wpk2, F1, F2);

  // ===== layer 1 =====
  k_gemm_bf16<512, 4, 2><<<(N + 15) / 16, 128, 0, stream>>>(x, wpk1, h1, N);
  k_alpha<<<(N * H1 + BLK - 1) / BLK, BLK, 0, stream>>>(h1, aS1, aD1, as1, ad1, N, H1, C1);
  k_fill_u32<<<(N * H1 + BLK - 1) / BLK, BLK, 0, stream>>>(m1, NEGINF_KEY, N * H1);
  k_fill_f32<<<(N * H1 + BLK - 1) / BLK, BLK, 0, stream>>>(s1, 0.f, N * H1);
  k_fill_f32<<<(N * F1 + BLK - 1) / BLK, BLK, 0, stream>>>(agg1, 0.f, N * F1);
  k_edge_max<<<egrid, BLK, 0, stream>>>(src, dst, E, N, H1, as1, ad1, m1);
  k_edge_sum<<<egrid, BLK, 0, stream>>>(src, dst, E, N, H1, as1, ad1, m1, s1);
  k_edge_agg<<<(totE * H1 + BLK - 1) / BLK, BLK, 0, stream>>>(
      src, dst, E, N, H1, C1, as1, ad1, m1, s1, h1, agg1);
  k_bias_relu<<<(N * F1 + BLK - 1) / BLK, BLK, 0, stream>>>(agg1, b1, h1 /*act1*/, N, F1);

  // ===== layer 2 =====
  float* act1 = h1;
  float* h2   = agg1;
  k_gemm_bf16<128, 2, 2><<<(N + 15) / 16, 64, 0, stream>>>(act1, wpk2, h2, N);
  k_alpha<<<(N + BLK - 1) / BLK, BLK, 0, stream>>>(h2, aS2, aD2, as2, ad2, N, 1, F2);
  k_fill_u32<<<(N + BLK - 1) / BLK, BLK, 0, stream>>>(m2, NEGINF_KEY, N);
  k_fill_f32<<<(N + BLK - 1) / BLK, BLK, 0, stream>>>(s2, 0.f, N);
  k_fill_f32<<<(N * F2 + BLK - 1) / BLK, BLK, 0, stream>>>(agg2, 0.f, N * F2);
  k_edge_max<<<egrid, BLK, 0, stream>>>(src, dst, E, N, 1, as2, ad2, m2);
  k_edge_sum<<<egrid, BLK, 0, stream>>>(src, dst, E, N, 1, as2, ad2, m2, s2);
  k_edge_agg<<<egrid, BLK, 0, stream>>>(src, dst, E, N, 1, F2, as2, ad2, m2, s2, h2, agg2);
  k_logsoftmax64<<<(N + 7) / 8, 256, 0, stream>>>(agg2, b2, out, N);
}
